// ContrastiveLoss_10350871183698
// MI455X (gfx1250) — matrix operand order repaired and checked by reading, not compiled, not run on hardware
//
#include <hip/hip_runtime.h>

typedef unsigned short u16;
typedef unsigned int u32;

typedef __attribute__((ext_vector_type(16))) __bf16 v16bf;
typedef __attribute__((ext_vector_type(8)))  float  v8f;
typedef int v4i __attribute__((vector_size(16)));   // matches builtin param type

#define NROWS 4096
#define DDIM  1024
#define INV_TEMP 20.0f
#define SHIFT 21.0f

// ---------- async global->LDS path (gfx1250), with compile-safe fallback ----
#if defined(__gfx1250__) && __has_builtin(__builtin_amdgcn_global_load_async_to_lds_b128)
#define HAVE_ASYNC_LDS 1
#else
#define HAVE_ASYNC_LDS 0
#endif

__device__ __forceinline__ void cp16_to_lds(void* l, const void* g) {
#if HAVE_ASYNC_LDS
    __builtin_amdgcn_global_load_async_to_lds_b128(
        (__attribute__((address_space(1))) v4i*)(g),
        (__attribute__((address_space(3))) v4i*)(l), 0, 0);
#else
    *reinterpret_cast<uint4*>(l) = *reinterpret_cast<const uint4*>(g);
#endif
}

__device__ __forceinline__ void wait_async_all() {
#if HAVE_ASYNC_LDS
#if __has_builtin(__builtin_amdgcn_s_wait_asynccnt)
    __builtin_amdgcn_s_wait_asynccnt(0);
#else
    asm volatile("s_wait_asynccnt 0x0" ::: "memory");
#endif
#endif
}

// ---------- helpers ----------
__device__ __forceinline__ u16 f2bf(float f) {
    u32 u = __float_as_uint(f);
    u32 r = (u + 0x7FFFu + ((u >> 16) & 1u)) >> 16;   // RNE
    return (u16)r;
}
__device__ __forceinline__ float bf2f(u16 h) {
    return __uint_as_float(((u32)h) << 16);
}

// ---------- kernel 1: row L2-normalize fp32 -> bf16 ----------
__global__ __launch_bounds__(256)
void cl_norm_kernel(const float* __restrict__ in, const float* __restrict__ tg,
                    const float* __restrict__ hn,
                    u16* __restrict__ xb, u16* __restrict__ tb, u16* __restrict__ hb) {
    const int row = blockIdx.x;
    const int which = blockIdx.y;
    const float* src = (which == 0) ? in : ((which == 1) ? tg : hn);
    u16*       dst  = (which == 0) ? xb : ((which == 1) ? tb : hb);

    const float4* s4 = reinterpret_cast<const float4*>(src + (size_t)row * DDIM);
    float4 v = s4[threadIdx.x];
    float ss = v.x * v.x + v.y * v.y + v.z * v.z + v.w * v.w;

    __shared__ float red[256];
    red[threadIdx.x] = ss;
    __syncthreads();
    for (int s = 128; s > 0; s >>= 1) {
        if (threadIdx.x < s) red[threadIdx.x] += red[threadIdx.x + s];
        __syncthreads();
    }
    float inv = 1.0f / fmaxf(sqrtf(red[0]), 1e-8f);

    uint2 o;
    o.x = (u32)f2bf(v.x * inv) | ((u32)f2bf(v.y * inv) << 16);
    o.y = (u32)f2bf(v.z * inv) | ((u32)f2bf(v.w * inv) << 16);
    reinterpret_cast<uint2*>(dst + (size_t)row * DDIM)[threadIdx.x] = o;
}

// ---------- kernel 2: diagonal positive logit + zero rowsum ----------
__global__ __launch_bounds__(256)
void cl_diag_kernel(const u16* __restrict__ xb, const u16* __restrict__ tb,
                    float* __restrict__ diag, float* __restrict__ rowsum) {
    const int row = blockIdx.x;
    const uint2 xa = reinterpret_cast<const uint2*>(xb + (size_t)row * DDIM)[threadIdx.x];
    const uint2 ta = reinterpret_cast<const uint2*>(tb + (size_t)row * DDIM)[threadIdx.x];
    float p = bf2f((u16)(xa.x & 0xFFFF)) * bf2f((u16)(ta.x & 0xFFFF))
            + bf2f((u16)(xa.x >> 16))    * bf2f((u16)(ta.x >> 16))
            + bf2f((u16)(xa.y & 0xFFFF)) * bf2f((u16)(ta.y & 0xFFFF))
            + bf2f((u16)(xa.y >> 16))    * bf2f((u16)(ta.y >> 16));

    __shared__ float red[256];
    red[threadIdx.x] = p;
    __syncthreads();
    for (int s = 128; s > 0; s >>= 1) {
        if (threadIdx.x < s) red[threadIdx.x] += red[threadIdx.x + s];
        __syncthreads();
    }
    if (threadIdx.x == 0) {
        diag[row]   = red[0];
        rowsum[row] = 0.0f;
    }
}

// ---------- kernel 3: bf16 WMMA GEMM tile + fused exp/rowsum ----------
// Block tile: 256 rows x 128 cols of the [N, 2N] logit matrix, BK=32,
// double-buffered LDS fed by async global->LDS copies. 8 waves, each
// owning a 64x64 region = 4x4 subtiles of v_wmma_f32_16x16x32_bf16
// (16 WMMA : 16 ds_load_b128 per K-step).
#define LDS_STRIDE_DW 20   // 80-byte padded row stride -> conflict-free b128 reads

__device__ __forceinline__ v16bf ld_frag(const u32* s, int dw0, int dw1) {
    union { uint4 u[2]; v16bf v; } t;
    t.u[0] = *reinterpret_cast<const uint4*>(s + dw0);
    t.u[1] = *reinterpret_cast<const uint4*>(s + dw1);
    return t.v;
}

__device__ __forceinline__ void stage_tiles(const u16* __restrict__ X,
                                            const u16* __restrict__ Y,
                                            int row0, int ycol0, int kt,
                                            u32* sA, u32* sB, int t) {
    // A tile: 256 rows x 64B = 1024 chunks of 16B; 4 per thread
    #pragma unroll
    for (int rep = 0; rep < 4; ++rep) {
        const int c = rep * 256 + t;
        const int row = c >> 2, off = c & 3;
        cp16_to_lds(&sA[row * LDS_STRIDE_DW + off * 4],
                    X + (size_t)(row0 + row) * DDIM + kt * 32 + off * 8);
    }
    // B tile: 128 rows x 64B = 512 chunks; 2 per thread
    #pragma unroll
    for (int rep = 0; rep < 2; ++rep) {
        const int c = rep * 256 + t;
        const int row = c >> 2, off = c & 3;
        cp16_to_lds(&sB[row * LDS_STRIDE_DW + off * 4],
                    Y + (size_t)(ycol0 + row) * DDIM + kt * 32 + off * 8);
    }
}

__global__ __launch_bounds__(256)
void cl_gemm_kernel(const u16* __restrict__ X, const u16* __restrict__ T,
                    const u16* __restrict__ H, float* __restrict__ rowsum) {
    __shared__ __align__(16) u32 sA[2][256 * LDS_STRIDE_DW];   // 2 x 20 KB
    __shared__ __align__(16) u32 sB[2][128 * LDS_STRIDE_DW];   // 2 x 10 KB

    const int row0 = blockIdx.y * 256;
    const int col0 = blockIdx.x * 128;          // global column in [0, 2N)
    const bool isNeg = (col0 >= NROWS);
    const u16* Y = isNeg ? H : T;
    const int ycol0 = isNeg ? (col0 - NROWS) : col0;

    const int t    = threadIdx.x;
    const int lane = t & 31;
    const int w    = t >> 5;
    const int m16  = lane & 15;
    const int hi   = lane >> 4;
    const int rbase = (w >> 1) * 64;            // wave row offset in tile
    const int cbase = (w & 1) * 64;             // wave col offset in tile

    v8f acc[4][4] = {};

    // prologue: fill buffer 0
    stage_tiles(X, Y, row0, ycol0, 0, sA[0], sB[0], t);
    wait_async_all();
    __syncthreads();

    for (int kt = 0; kt < DDIM / 32; ++kt) {
        const int cur = kt & 1;
        if (kt + 1 < DDIM / 32)   // async-prefetch next K-slice into other buffer
            stage_tiles(X, Y, row0, ycol0, kt + 1, sA[cur ^ 1], sB[cur ^ 1], t);

        const u32* cA = sA[cur];
        const u32* cB = sB[cur];
        v16bf af[4], bfr[4];
        #pragma unroll
        for (int i = 0; i < 4; ++i) {
            const int r = rbase + i * 16 + m16;
            // A 16x32 bf16: VGPR0-3 hold K=hi*8..+7, VGPR4-7 hold K=hi*8+16..+23
            af[i] = ld_frag(cA, r * LDS_STRIDE_DW + hi * 4,
                                r * LDS_STRIDE_DW + hi * 4 + 8);
        }
        #pragma unroll
        for (int j = 0; j < 4; ++j) {
            const int yr = cbase + j * 16 + m16;
            // B 32x16 bf16: VGPR0-3 hold K=hi*8..+7, VGPR4-7 hold K=hi*8+16..+23
            bfr[j] = ld_frag(cB, yr * LDS_STRIDE_DW + hi * 4,
                                 yr * LDS_STRIDE_DW + hi * 4 + 8);
        }
        #pragma unroll
        for (int i = 0; i < 4; ++i)
            #pragma unroll
            for (int j = 0; j < 4; ++j)
                acc[i][j] = __builtin_amdgcn_wmma_f32_16x16x32_bf16(
                    false, af[i], false, bfr[j], (short)0, acc[i][j], false, false);

        if (kt + 1 < DDIM / 32) {
            wait_async_all();   // next buffer landed in LDS
            __syncthreads();    // and everyone finished reading current buffer
        }
    }

    // Epilogue: logits = sim*20 (+1 on hard-neg diagonal); accumulate
    // exp(logit - 21) per row via half-wave shfl reduce + one atomic.
    #pragma unroll
    for (int i = 0; i < 4; ++i) {
        #pragma unroll
        for (int v = 0; v < 8; ++v) {
            const int r = row0 + rbase + i * 16 + hi * 8 + v;   // C/D: M = v + 8*hi
            float p = 0.0f;
            #pragma unroll
            for (int j = 0; j < 4; ++j) {
                float logit = acc[i][j][v] * INV_TEMP;
                const int yc = ycol0 + cbase + j * 16 + m16;    // C/D: N = lane%16
                if (isNeg && yc == r) logit += 1.0f;            // HARD_NEG_WEIGHT
                p += __expf(logit - SHIFT);
            }
            p += __shfl_xor(p, 1, 32);
            p += __shfl_xor(p, 2, 32);
            p += __shfl_xor(p, 4, 32);
            p += __shfl_xor(p, 8, 32);
            if (m16 == 0) atomicAdd(&rowsum[r], p);
        }
    }
}

// ---------- kernel 4: final scalar loss ----------
__global__ __launch_bounds__(256)
void cl_loss_kernel(const float* __restrict__ rowsum, const float* __restrict__ diag,
                    float* __restrict__ out) {
    float acc = 0.0f;
    for (int i = threadIdx.x; i < NROWS; i += 256)
        acc += SHIFT + __logf(rowsum[i]) - diag[i] * INV_TEMP;
    __shared__ float red[256];
    red[threadIdx.x] = acc;
    __syncthreads();
    for (int s = 128; s > 0; s >>= 1) {
        if (threadIdx.x < s) red[threadIdx.x] += red[threadIdx.x + s];
        __syncthreads();
    }
    if (threadIdx.x == 0) out[0] = red[0] * (1.0f / (float)NROWS);
}

extern "C" void kernel_launch(void* const* d_in, const int* in_sizes, int n_in,
                              void* d_out, int out_size, void* d_ws, size_t ws_size,
                              hipStream_t stream) {
    const float* in = (const float*)d_in[0];
    const float* tg = (const float*)d_in[1];
    const float* hn = (const float*)d_in[2];
    float* out = (float*)d_out;

    // workspace: 3 bf16 matrices (24 MB) + diag + rowsum (32 KB)
    u16* xb = (u16*)d_ws;
    u16* tb = xb + (size_t)NROWS * DDIM;
    u16* hb = tb + (size_t)NROWS * DDIM;
    float* diag   = (float*)(hb + (size_t)NROWS * DDIM);
    float* rowsum = diag + NROWS;

    cl_norm_kernel<<<dim3(NROWS, 3), 256, 0, stream>>>(in, tg, hn, xb, tb, hb);
    cl_diag_kernel<<<NROWS, 256, 0, stream>>>(xb, tb, diag, rowsum);
    cl_gemm_kernel<<<dim3(2 * NROWS / 128, NROWS / 256), 256, 0, stream>>>(xb, tb, hb, rowsum);
    cl_loss_kernel<<<1, 256, 0, stream>>>(rowsum, diag, out);
}
